// PointSorterNO_old_20083267076193
// MI455X (gfx1250) — matrix operand order repaired and checked by reading, not compiled, not run on hardware
//
#include <hip/hip_runtime.h>
#include <hip/hip_bf16.h>

#define NPTS 131072
#define KNBR 16
#define HD   64
#define KP   96      // K padded from 67 to 3 wmma chunks of 32
#define EPS  1e-5f

typedef __attribute__((ext_vector_type(16))) __bf16 bf16x16;
typedef __attribute__((ext_vector_type(8)))  float  f32x8;

union Frag {
    bf16x16 v;
    uint4   q[2];
};

static __device__ __forceinline__ unsigned short f2bf(float f) {
    union { float f; unsigned u; } x{f};
    unsigned r = x.u + 0x7FFFu + ((x.u >> 16) & 1u);   // round-to-nearest-even
    return (unsigned short)(r >> 16);
}
static __device__ __forceinline__ float bf2f(unsigned short h) {
    union { unsigned u; float f; } x;
    x.u = ((unsigned)h) << 16;
    return x.f;
}
static __device__ __forceinline__ float gelu_exact(float x) {
    return 0.5f * x * (1.0f + erff(x * 0.70710678118654752f));
}

// ---------------------------------------------------------------------------
// Lift: h = [coord | feat] @ lift_w + lift_b    (N,9)@(9,64) -> bf16 table
// ---------------------------------------------------------------------------
__global__ __launch_bounds__(256) void lift_kernel(
    const float* __restrict__ coord, const float* __restrict__ feat,
    const float* __restrict__ w, const float* __restrict__ b,
    unsigned short* __restrict__ h)
{
    int i = blockIdx.x * 256 + threadIdx.x;
    if (i >= NPTS) return;
    float in[9];
    in[0] = coord[i*3+0]; in[1] = coord[i*3+1]; in[2] = coord[i*3+2];
    #pragma unroll
    for (int c = 0; c < 6; ++c) in[3+c] = feat[i*6+c];
    #pragma unroll 4
    for (int n = 0; n < HD; ++n) {
        float a = b[n];
        #pragma unroll
        for (int t = 0; t < 9; ++t) a += in[t] * w[t*HD + n];
        h[(size_t)i*HD + n] = f2bf(a);
    }
}

// ---------------------------------------------------------------------------
// Skip GEMM: skip = h @ sw + sb   (N,64)@(64,64), one wave per 16 points.
// A fragments loaded directly from the global bf16 h table (rows are 128B
// aligned; chunk offsets are 16B multiples). B staged transposed in LDS.
// ---------------------------------------------------------------------------
__global__ __launch_bounds__(256) void skip_kernel(
    const unsigned short* __restrict__ h,
    const float* __restrict__ sw, const float* __restrict__ sb,
    unsigned short* __restrict__ skipT)
{
    __shared__ __align__(16) unsigned short sSW[HD * HD];   // [n][k], 8 KB

    const int tid  = threadIdx.x;
    const int wave = tid >> 5;
    const int lane = tid & 31;
    const int r    = lane & 15;
    const int hi   = lane >> 4;

    for (int e = tid; e < HD * HD; e += 256) {
        int n = e >> 6, k = e & 63;
        sSW[e] = f2bf(sw[k*HD + n]);
    }
    __syncthreads();

    const int m0 = (blockIdx.x * 8 + wave) * 16;
    const unsigned short* arow = h + (size_t)(m0 + r) * HD;

    f32x8 acc[4];
    #pragma unroll
    for (int t = 0; t < 4; ++t) {
        float bn = sb[t*16 + r];
        #pragma unroll
        for (int q = 0; q < 8; ++q) acc[t][q] = bn;
    }
    #pragma unroll
    for (int c = 0; c < 2; ++c) {
        Frag a;
        a.q[0] = *(const uint4*)(arow + 32*c + 8*hi);
        a.q[1] = *(const uint4*)(arow + 32*c + 16 + 8*hi);
        #pragma unroll
        for (int t = 0; t < 4; ++t) {
            const unsigned short* bcol = &sSW[(t*16 + r) * HD + 32*c];
            Frag b;
            b.q[0] = *(const uint4*)(bcol + 8*hi);
            b.q[1] = *(const uint4*)(bcol + 16 + 8*hi);
            acc[t] = __builtin_amdgcn_wmma_f32_16x16x32_bf16(
                false, a.v, false, b.v, (short)0, acc[t], false, false);
        }
    }
    #pragma unroll
    for (int t = 0; t < 4; ++t)
        #pragma unroll
        for (int q = 0; q < 8; ++q)
            skipT[(size_t)(m0 + q + 8*hi) * HD + (t*16 + r)] = f2bf(acc[t][q]);
}

// ---------------------------------------------------------------------------
// One GNO layer + ReLU + LayerNorm.  One wave32 per point.
// kin K-order is permuted to [h | rel | pad] so gathered rows copy as b128s;
// W1 LDS staging applies the matching row permutation.
// ---------------------------------------------------------------------------
__global__ __launch_bounds__(256) void gno_layer_kernel(
    const float* __restrict__ coord, const int* __restrict__ idx,
    const unsigned short* __restrict__ h_in, unsigned short* __restrict__ h_out,
    const unsigned short* __restrict__ skipT,
    const float* __restrict__ w1, const float* __restrict__ b1,
    const float* __restrict__ w2, const float* __restrict__ b2,
    const float* __restrict__ lng, const float* __restrict__ lnb)
{
    __shared__ __align__(16) unsigned short sW1[HD * KP];       // [n][k'] permuted, 12 KB
    __shared__ __align__(16) unsigned short sW2[HD * HD];       // [n][k],            8 KB
    __shared__ __align__(16) unsigned short sKin[8][KNBR * KP]; // per-wave A tile,  24 KB
    __shared__ __align__(16) unsigned short sHid[8][KNBR * HD]; // per-wave hid,     16 KB

    const int tid  = threadIdx.x;
    const int wave = tid >> 5;
    const int lane = tid & 31;
    const int r    = lane & 15;   // row / column within 16
    const int hi   = lane >> 4;   // half-wave select

    // ---- stage transposed bf16 weights (K-permuted for W1) ----
    for (int e = tid; e < HD * KP; e += 256) {
        int n = e / KP, k = e - n * KP;      // k is permuted index
        float v;
        if      (k < HD)     v = w1[(3 + k) * HD + n];   // h-feature rows
        else if (k < HD + 3) v = w1[(k - HD) * HD + n];  // rel rows
        else                 v = 0.0f;                   // pad
        sW1[e] = f2bf(v);
    }
    for (int e = tid; e < HD * HD; e += 256) {
        int n = e >> 6, k = e & 63;
        sW2[e] = f2bf(w2[k*HD + n]);
    }

    const int p = blockIdx.x * 8 + wave;

    // ---- build kin = [h[idx] | rel | 0-pad] for 16 neighbors (vectorized) ----
    {
        int j = idx[p*KNBR + r];                       // both halves same neighbor
        unsigned short* row = &sKin[wave][r * KP];
        const uint4* hsrc = (const uint4*)(h_in + (size_t)j * HD);
        #pragma unroll
        for (int u = 0; u < 4; ++u)                    // 32 halves per half-wave
            *(uint4*)(row + hi*32 + u*8) = hsrc[hi*4 + u];
        if (hi == 0) {
            uint4 z; z.x = z.y = z.z = z.w = 0u;
            #pragma unroll
            for (int u = 0; u < 4; ++u)                // zero slots 64..95
                *(uint4*)(row + HD + u*8) = z;
            float cx = coord[p*3+0], cy = coord[p*3+1], cz = coord[p*3+2];
            row[HD+0] = f2bf(coord[j*3+0] - cx);
            row[HD+1] = f2bf(coord[j*3+1] - cy);
            row[HD+2] = f2bf(coord[j*3+2] - cz);
        }
    }
    __syncthreads();

    // ---- GEMM1: 3 k-chunks x 4 n-tiles of v_wmma_f32_16x16x32_bf16 ----
    f32x8 acc[4];
    #pragma unroll
    for (int t = 0; t < 4; ++t) {
        float bn = b1[t*16 + r];          // bias folded into accumulator
        #pragma unroll
        for (int q = 0; q < 8; ++q) acc[t][q] = bn;
    }
    const unsigned short* arow = &sKin[wave][r * KP];
    #pragma unroll
    for (int c = 0; c < 3; ++c) {
        Frag a;
        a.q[0] = *(const uint4*)(arow + 32*c + 8*hi);        // K {0-7}/{8-15}
        a.q[1] = *(const uint4*)(arow + 32*c + 16 + 8*hi);   // K {16-23}/{24-31}
        #pragma unroll
        for (int t = 0; t < 4; ++t) {
            const unsigned short* bcol = &sW1[(t*16 + r) * KP + 32*c];
            Frag b;
            b.q[0] = *(const uint4*)(bcol + 8*hi);
            b.q[1] = *(const uint4*)(bcol + 16 + 8*hi);
            acc[t] = __builtin_amdgcn_wmma_f32_16x16x32_bf16(
                false, a.v, false, b.v, (short)0, acc[t], false, false);
        }
    }

    // ---- exact GELU, spill hid tile to LDS as bf16 (A-layout round trip) ----
    #pragma unroll
    for (int t = 0; t < 4; ++t) {
        #pragma unroll
        for (int q = 0; q < 8; ++q) {
            float g = gelu_exact(acc[t][q]);
            sHid[wave][(q + 8*hi) * HD + (t*16 + r)] = f2bf(g);  // [M][N]
        }
    }
    __syncthreads();

    // ---- GEMM2: 2 k-chunks x 4 n-tiles ----
    f32x8 acc2[4];
    #pragma unroll
    for (int t = 0; t < 4; ++t)
        #pragma unroll
        for (int q = 0; q < 8; ++q) acc2[t][q] = 0.0f;

    const unsigned short* a2row = &sHid[wave][r * HD];
    #pragma unroll
    for (int c = 0; c < 2; ++c) {
        Frag a;
        a.q[0] = *(const uint4*)(a2row + 32*c + 8*hi);
        a.q[1] = *(const uint4*)(a2row + 32*c + 16 + 8*hi);
        #pragma unroll
        for (int t = 0; t < 4; ++t) {
            const unsigned short* bcol = &sW2[(t*16 + r) * HD + 32*c];
            Frag b;
            b.q[0] = *(const uint4*)(bcol + 8*hi);
            b.q[1] = *(const uint4*)(bcol + 16 + 8*hi);
            acc2[t] = __builtin_amdgcn_wmma_f32_16x16x32_bf16(
                false, a.v, false, b.v, (short)0, acc2[t], false, false);
        }
    }

    // ---- mean over 16 neighbors + bias + precomputed skip + relu ----
    float outv[4];
    #pragma unroll
    for (int t = 0; t < 4; ++t) {
        float s = 0.0f;
        #pragma unroll
        for (int q = 0; q < 8; ++q) s += acc2[t][q];     // partial rows M=8*hi..8*hi+7
        s += __shfl_xor(s, 16);                          // full 16-row column sum
        int n = t*16 + r;
        float v = s * (1.0f / 16.0f) + b2[n]
                + bf2f(skipT[(size_t)p*HD + n]);         // h@sw + sb (WMMA'd)
        outv[t] = v > 0.0f ? v : 0.0f;                   // relu
    }

    // ---- layernorm over 64 channels (each channel duplicated on lane^16) ----
    float lsum = outv[0] + outv[1] + outv[2] + outv[3];
    #pragma unroll
    for (int o = 1; o < 32; o <<= 1) lsum += __shfl_xor(lsum, o);
    float mean = lsum * (1.0f / 128.0f);                 // /64, dedup factor 2
    float lvar = 0.0f;
    #pragma unroll
    for (int t = 0; t < 4; ++t) { float d = outv[t] - mean; lvar += d * d; }
    #pragma unroll
    for (int o = 1; o < 32; o <<= 1) lvar += __shfl_xor(lvar, o);
    float rstd = rsqrtf(lvar * (1.0f / 128.0f) + EPS);

    if (hi == 0) {
        #pragma unroll
        for (int t = 0; t < 4; ++t) {
            int n = t*16 + r;
            h_out[(size_t)p*HD + n] = f2bf((outv[t] - mean) * rstd * lng[n] + lnb[n]);
        }
    }
}

// ---------------------------------------------------------------------------
// Score head (WMMA): hid = gelu(h @ sh_w1 + b1), s = hid @ sh_w2 + b2,
// score = sigmoid(s).  One wave per 16 points, 4 WMMAs; the 32->1 dot is a
// half-wave shuffle reduction per output row.
// ---------------------------------------------------------------------------
__global__ __launch_bounds__(256) void head_kernel(
    const unsigned short* __restrict__ h,
    const float* __restrict__ w1, const float* __restrict__ b1,
    const float* __restrict__ w2, const float* __restrict__ b2,
    float* __restrict__ scores, float* __restrict__ keys, int* __restrict__ vals)
{
    __shared__ __align__(16) unsigned short sW1T[32 * HD];  // [n][k], 4 KB

    const int tid  = threadIdx.x;
    const int wave = tid >> 5;
    const int lane = tid & 31;
    const int r    = lane & 15;
    const int hi   = lane >> 4;

    for (int e = tid; e < 32 * HD; e += 256) {
        int n = e >> 6, k = e & 63;
        sW1T[e] = f2bf(w1[k*32 + n]);
    }
    __syncthreads();

    const int m0 = (blockIdx.x * 8 + wave) * 16;
    const unsigned short* arow = h + (size_t)(m0 + r) * HD;

    f32x8 acc[2];
    #pragma unroll
    for (int t = 0; t < 2; ++t) {
        float bn = b1[t*16 + r];
        #pragma unroll
        for (int q = 0; q < 8; ++q) acc[t][q] = bn;
    }
    #pragma unroll
    for (int c = 0; c < 2; ++c) {
        Frag a;
        a.q[0] = *(const uint4*)(arow + 32*c + 8*hi);
        a.q[1] = *(const uint4*)(arow + 32*c + 16 + 8*hi);
        #pragma unroll
        for (int t = 0; t < 2; ++t) {
            const unsigned short* bcol = &sW1T[(t*16 + r) * HD + 32*c];
            Frag b;
            b.q[0] = *(const uint4*)(bcol + 8*hi);
            b.q[1] = *(const uint4*)(bcol + 16 + 8*hi);
            acc[t] = __builtin_amdgcn_wmma_f32_16x16x32_bf16(
                false, a.v, false, b.v, (short)0, acc[t], false, false);
        }
    }

    // gelu * w2[n], then sum the 32 hidden channels per row (half-wave reduce)
    float w2a = w2[r], w2b = w2[16 + r];
    float sc2 = b2[0];
    #pragma unroll
    for (int q = 0; q < 8; ++q) {
        float s = gelu_exact(acc[0][q]) * w2a + gelu_exact(acc[1][q]) * w2b;
        #pragma unroll
        for (int o = 1; o < 16; o <<= 1) s += __shfl_xor(s, o);  // stays in half
        if (r == 0) {
            int i = m0 + q + 8*hi;                     // row M = q + 8*hi
            float sc = 1.0f / (1.0f + expf(-(s + sc2)));
            scores[i] = sc;
            keys[i]   = sc;
            vals[i]   = i;
        }
    }
}

// ---------------------------------------------------------------------------
// Global bitonic sort (ascending) on (key,val) pairs; batch offsets are zero.
// ---------------------------------------------------------------------------
__global__ __launch_bounds__(256) void bitonic_step(
    float* __restrict__ key, int* __restrict__ val, int j, int k)
{
    unsigned i = blockIdx.x * 256u + threadIdx.x;
    unsigned ixj = i ^ (unsigned)j;
    if (ixj > i) {
        float ki = key[i], kj = key[ixj];
        bool up = ((i & (unsigned)k) == 0u);
        if ((ki > kj) == up) {
            key[i] = kj; key[ixj] = ki;
            int vi = val[i]; val[i] = val[ixj]; val[ixj] = vi;
        }
    }
}

__global__ __launch_bounds__(256) void finalize_kernel(
    const int* __restrict__ order, int* __restrict__ out_orders, int* __restrict__ out_inv)
{
    int i = blockIdx.x * 256 + threadIdx.x;
    if (i >= NPTS) return;
    int o = order[i];
    out_orders[i] = o;     // orders[0, i]
    out_inv[o]    = i;     // inverses[0, orders[i]] = i
}

// ---------------------------------------------------------------------------
extern "C" void kernel_launch(void* const* d_in, const int* in_sizes, int n_in,
                              void* d_out, int out_size, void* d_ws, size_t ws_size,
                              hipStream_t stream)
{
    const float* coord  = (const float*)d_in[0];
    const float* feat   = (const float*)d_in[1];
    const int*   idx    = (const int*)  d_in[2];
    // d_in[3] = batch (all zeros -> offsets vanish)
    const float* lift_w = (const float*)d_in[4];
    const float* lift_b = (const float*)d_in[5];
    const float* g1_w1  = (const float*)d_in[6];
    const float* g1_b1  = (const float*)d_in[7];
    const float* g1_w2  = (const float*)d_in[8];
    const float* g1_b2  = (const float*)d_in[9];
    const float* g1_sw  = (const float*)d_in[10];
    const float* g1_sb  = (const float*)d_in[11];
    const float* g2_w1  = (const float*)d_in[12];
    const float* g2_b1  = (const float*)d_in[13];
    const float* g2_w2  = (const float*)d_in[14];
    const float* g2_b2  = (const float*)d_in[15];
    const float* g2_sw  = (const float*)d_in[16];
    const float* g2_sb  = (const float*)d_in[17];
    const float* ln1_g  = (const float*)d_in[18];
    const float* ln1_b  = (const float*)d_in[19];
    const float* ln2_g  = (const float*)d_in[20];
    const float* ln2_b  = (const float*)d_in[21];
    const float* sh_w1  = (const float*)d_in[22];
    const float* sh_b1  = (const float*)d_in[23];
    const float* sh_w2  = (const float*)d_in[24];
    const float* sh_b2  = (const float*)d_in[25];

    // workspace carve-up
    char* ws = (char*)d_ws;
    unsigned short* hA    = (unsigned short*)ws;  ws += (size_t)NPTS * HD * sizeof(unsigned short);
    unsigned short* hB    = (unsigned short*)ws;  ws += (size_t)NPTS * HD * sizeof(unsigned short);
    unsigned short* skipT = (unsigned short*)ws;  ws += (size_t)NPTS * HD * sizeof(unsigned short);
    float* keys           = (float*)ws;           ws += (size_t)NPTS * sizeof(float);
    int*   vals           = (int*)ws;

    float* out_scores = (float*)d_out;
    int*   out_orders = (int*)((float*)d_out + NPTS);
    int*   out_inv    = out_orders + NPTS;

    const int TPB  = 256;
    const int nblk = NPTS / TPB;          // 512
    const int gblk = NPTS / 8;            // 16384 (1 point per wave)
    const int wblk = NPTS / 128;          // 1024  (16 points per wave)

    lift_kernel<<<nblk, TPB, 0, stream>>>(coord, feat, lift_w, lift_b, hA);

    skip_kernel<<<wblk, TPB, 0, stream>>>(hA, g1_sw, g1_sb, skipT);
    gno_layer_kernel<<<gblk, TPB, 0, stream>>>(
        coord, idx, hA, hB, skipT, g1_w1, g1_b1, g1_w2, g1_b2, ln1_g, ln1_b);

    skip_kernel<<<wblk, TPB, 0, stream>>>(hB, g2_sw, g2_sb, skipT);
    gno_layer_kernel<<<gblk, TPB, 0, stream>>>(
        coord, idx, hB, hA, skipT, g2_w1, g2_b1, g2_w2, g2_b2, ln2_g, ln2_b);

    head_kernel<<<wblk, TPB, 0, stream>>>(hA, sh_w1, sh_b1, sh_w2, sh_b2,
                                          out_scores, keys, vals);

    for (int k = 2; k <= NPTS; k <<= 1)
        for (int j = k >> 1; j > 0; j >>= 1)
            bitonic_step<<<nblk, TPB, 0, stream>>>(keys, vals, j, k);

    finalize_kernel<<<nblk, TPB, 0, stream>>>(vals, out_orders, out_inv);
}
